// GCN_31250182045755
// MI455X (gfx1250) — matrix operand order repaired and checked by reading, not compiled, not run on hardware
//
#include <hip/hip_runtime.h>
#include <hip/hip_bf16.h>

// ---------------------------------------------------------------------------
// GCN forward for MI455X (gfx1250, wave32).
//   per layer:  H = X @ W            (bf16 WMMA, f32 accumulate)
//               OUT[col] += norm*H[row]   (edge-parallel, f32 atomics in L2)
//               OUT = relu(OUT + b)       (elementwise; last layer no relu)
// ---------------------------------------------------------------------------

typedef __attribute__((ext_vector_type(16))) __bf16 v16bf;
typedef __attribute__((ext_vector_type(8)))  float  v8f;

__device__ __forceinline__ unsigned short f32_to_bf16_rne(float f) {
    unsigned int u = __float_as_uint(f);
    unsigned int r = 0x7FFFu + ((u >> 16) & 1u);   // round-to-nearest-even
    u += r;
    return (unsigned short)(u >> 16);
}

// ------------------------------ utility kernels ----------------------------

__global__ void k_zero_f32(float* __restrict__ p, int n) {
    int i = blockIdx.x * blockDim.x + threadIdx.x;
    if (i < n) p[i] = 0.0f;
}

__global__ void k_cvt_f32_bf16(const float* __restrict__ src,
                               unsigned short* __restrict__ dst, int n) {
    int i = blockIdx.x * blockDim.x + threadIdx.x;
    if (i < n) dst[i] = f32_to_bf16_rne(src[i]);
}

// deg[col[e]] += w[e]
__global__ void k_degree(const int* __restrict__ cols,
                         const float* __restrict__ w,
                         float* __restrict__ deg, int E) {
    int e = blockIdx.x * blockDim.x + threadIdx.x;
    if (e < E) atomicAdd(&deg[cols[e]], w[e]);
}

__global__ void k_dinv_sqrt(const float* __restrict__ deg,
                            float* __restrict__ dinv, int N) {
    int i = blockIdx.x * blockDim.x + threadIdx.x;
    if (i < N) {
        float d = deg[i];
        dinv[i] = (d > 0.0f) ? rsqrtf(fmaxf(d, 1e-30f)) : 0.0f;
    }
}

__global__ void k_norm(const int* __restrict__ rows,
                       const int* __restrict__ cols,
                       const float* __restrict__ w,
                       const float* __restrict__ dinv,
                       float* __restrict__ norm, int E) {
    int e = blockIdx.x * blockDim.x + threadIdx.x;
    if (e < E) norm[e] = dinv[rows[e]] * w[e] * dinv[cols[e]];
}

// X[i] = act(X[i] + b[i % F])  (F is a power of two here: 128 / 64)
__global__ void k_bias_act(float* __restrict__ x, const float* __restrict__ b,
                           int n, int F, int do_relu) {
    int i = blockIdx.x * blockDim.x + threadIdx.x;
    if (i < n) {
        float v = x[i] + b[i & (F - 1)];
        if (do_relu) v = fmaxf(v, 0.0f);
        x[i] = v;
    }
}

// ------------------------- B-matrix fragment pre-pack ----------------------
// Packed layout (per 16x16 col-tile ct, per K-step ks of 32):
//   frag[(ct*ksteps + ks)*512 + lane*16 + e]
//   lane < 16 : column n = ct*16 + lane,     K = ks*32 + e        (e = 0..15)
//   lane >=16 : column n = ct*16 + lane-16,  K = ks*32 + 16 + e
// Contiguous k order (not AMD's table order); the GEMM loads A frags in the same order.

__global__ void k_pack_w(const float* __restrict__ W, // [K][Ncols] row-major
                         unsigned short* __restrict__ Bp,
                         int K, int Ncols) {
    int ksteps = K >> 5;
    int total  = (Ncols >> 4) * ksteps * 512;
    int idx = blockIdx.x * blockDim.x + threadIdx.x;
    if (idx >= total) return;
    int e    = idx & 15;
    int lane = (idx >> 4) & 31;
    int rem  = idx >> 9;            // ct*ksteps + ks
    int ks   = rem % ksteps;
    int ct   = rem / ksteps;
    int n = (ct << 4) + (lane & 15);
    int k = (ks << 5) + ((lane >> 4) << 4) + e;
    Bp[idx] = f32_to_bf16_rne(W[(size_t)k * Ncols + n]);
}

// ------------------------------- WMMA GEMM ---------------------------------
// One wave computes one 16x16 tile of C = A(bf16) x B(bf16) with f32 acc.
// A: row-major bf16 [M][K].  Bp: packed fragments (see above).  C: f32 [M][Nc].

__global__ void k_gemm_wmma_bf16(const unsigned short* __restrict__ A,
                                 const unsigned short* __restrict__ Bp,
                                 float* __restrict__ C,
                                 int M, int K, int Ncols, int totalWaves) {
    int wave = (int)((blockIdx.x * blockDim.x + threadIdx.x) >> 5);
    if (wave >= totalWaves) return;               // wave-uniform: EXEC stays full
    int lane = threadIdx.x & 31;
    int nct  = Ncols >> 4;
    int mt   = wave / nct;
    int ct   = wave - mt * nct;
    int m0   = mt << 4;
    int hi   = lane >> 4;                         // half-wave select
    int mrow = m0 + (lane & 15);
    int ksteps = K >> 5;

    const unsigned short* arow  = A  + (size_t)mrow * K;
    const unsigned short* bbase = Bp + ((size_t)ct * ksteps << 9);

    v8f acc = {};
    for (int ks = 0; ks < ksteps; ++ks) {
        int k0 = ks << 5;
        union { uint4 q[2]; v16bf v; } ua, ub;
        // A frag: 16 contiguous K per lane, K = k0 + 16*hi + e (matches packed B)
        ua.q[0] = *(const uint4*)(arow + k0 + hi * 16);
        ua.q[1] = *(const uint4*)(arow + k0 + hi * 16 + 8);
        // B frag: pre-packed, fully contiguous per lane
        const unsigned short* bf = bbase + (ks << 9) + (lane << 4);
        ub.q[0] = *(const uint4*)(bf);
        ub.q[1] = *(const uint4*)(bf + 8);
        acc = __builtin_amdgcn_wmma_f32_16x16x32_bf16(
                  false, ua.v, false, ub.v, (short)0, acc, false, false);
    }

    // C/D 16x16 f32 layout: VGPR r -> row m0 + r + 8*hi, col = ct*16 + (lane&15)
    int col   = (ct << 4) + (lane & 15);
    int rbase = m0 + hi * 8;
#pragma unroll
    for (int r = 0; r < 8; ++r)
        C[(size_t)(rbase + r) * Ncols + col] = acc[r];
}

// ---------------------------- edge scatter-add -----------------------------
// thread t -> (edge e, 4-feature chunk).  Consecutive threads walk features of
// the same edge -> coalesced float4 gather of H[row]; atomics land in L2
// (25.6MB accumulator << 192MB L2).

__global__ void k_scatter(const float* __restrict__ H,
                          const int* __restrict__ rows,
                          const int* __restrict__ cols,
                          const float* __restrict__ norm,
                          float* __restrict__ OUT,
                          int E, int F) {
    int t = blockIdx.x * blockDim.x + threadIdx.x;
    int fchunks = F >> 2;
    int e  = t / fchunks;
    if (e >= E) return;
    int fc = (t - e * fchunks) << 2;
    float nv = norm[e];
    const float4 h4 = *(const float4*)(H + (size_t)rows[e] * F + fc);
    float* o = OUT + (size_t)cols[e] * F + fc;
    atomicAdd(o + 0, nv * h4.x);
    atomicAdd(o + 1, nv * h4.y);
    atomicAdd(o + 2, nv * h4.z);
    atomicAdd(o + 3, nv * h4.w);
}

// ------------------------------- launcher ----------------------------------

static inline int cdiv(long long a, int b) { return (int)((a + b - 1) / b); }

extern "C" void kernel_launch(void* const* d_in, const int* in_sizes, int n_in,
                              void* d_out, int out_size, void* d_ws, size_t ws_size,
                              hipStream_t stream) {
    const float* x  = (const float*)d_in[0];
    const int*   ei = (const int*)  d_in[1];
    const float* ew = (const float*)d_in[2];
    const float* W1 = (const float*)d_in[3];
    const float* b1 = (const float*)d_in[4];
    const float* W2 = (const float*)d_in[5];
    const float* b2 = (const float*)d_in[6];
    const float* W3 = (const float*)d_in[7];
    const float* b3 = (const float*)d_in[8];

    const int dh   = in_sizes[4];            // 128
    const int dout = in_sizes[8];            // 64
    const int din  = in_sizes[3] / dh;       // 128
    const int N    = in_sizes[0] / din;      // 50000
    const int E    = in_sizes[2];            // 800000
    const int* rows = ei;
    const int* cols = ei + E;

    // ---- workspace layout (256B aligned slabs) ----
    char* wp = (char*)d_ws;
    auto alloc = [&](size_t bytes) -> void* {
        void* p = wp; wp += (bytes + 255) & ~(size_t)255; return p;
    };
    float*          deg  = (float*)         alloc((size_t)N * 4);
    float*          dinv = (float*)         alloc((size_t)N * 4);
    float*          nrm  = (float*)         alloc((size_t)E * 4);
    unsigned short* Xb   = (unsigned short*)alloc((size_t)N * din * 2);
    float*          Hg   = (float*)         alloc((size_t)N * dh * 4);
    float*          ACC  = (float*)         alloc((size_t)N * dh * 4);
    unsigned short* Wp   = (unsigned short*)alloc((size_t)din * dh * 2);

    const int T = 256;

    // ---- normalization: deg -> deg^-1/2 -> per-edge norm ----
    k_zero_f32  <<<cdiv(N, T), T, 0, stream>>>(deg, N);
    k_degree    <<<cdiv(E, T), T, 0, stream>>>(cols, ew, deg, E);
    k_dinv_sqrt <<<cdiv(N, T), T, 0, stream>>>(deg, dinv, N);
    k_norm      <<<cdiv(E, T), T, 0, stream>>>(rows, cols, ew, dinv, nrm, E);

    auto layer = [&](const unsigned short* Abf, const float* W, const float* b,
                     int K, int F, float* outBuf, int do_relu) {
        int packElems = (F >> 4) * (K >> 5) * 512;
        k_pack_w<<<cdiv(packElems, T), T, 0, stream>>>(W, Wp, K, F);
        int waves = (N >> 4) * (F >> 4);
        k_gemm_wmma_bf16<<<cdiv((long long)waves * 32, T), T, 0, stream>>>(
            Abf, Wp, Hg, N, K, F, waves);
        k_zero_f32<<<cdiv((long long)N * F, T), T, 0, stream>>>(outBuf, N * F);
        long long sthreads = (long long)E * (F >> 2);
        k_scatter<<<cdiv(sthreads, T), T, 0, stream>>>(Hg, rows, cols, nrm,
                                                       outBuf, E, F);
        k_bias_act<<<cdiv((long long)N * F, T), T, 0, stream>>>(outBuf, b,
                                                                N * F, F, do_relu);
    };

    // ---- layer 1 ----
    k_cvt_f32_bf16<<<cdiv((long long)N * din, T), T, 0, stream>>>(x, Xb, N * din);
    layer(Xb, W1, b1, din, dh, ACC, 1);

    // ---- layer 2 ----
    k_cvt_f32_bf16<<<cdiv((long long)N * dh, T), T, 0, stream>>>(ACC, Xb, N * dh);
    layer(Xb, W2, b2, dh, dh, ACC, 1);

    // ---- layer 3 (into d_out, no relu) ----
    k_cvt_f32_bf16<<<cdiv((long long)N * dh, T), T, 0, stream>>>(ACC, Xb, N * dh);
    layer(Xb, W3, b3, dh, dout, (float*)d_out, 0);
}
